// SparseQuery_10874857193582
// MI455X (gfx1250) — matrix operand order repaired and checked by reading, not compiled, not run on hardware
//
#include <hip/hip_runtime.h>
#include <hip/hip_bf16.h>

#define IN_F   2048
#define HEADS  16
#define HD     128
#define TOPK   4
#define HIDDEN 256
#define NTOK   8192   // B*S = 4*2048
#define KCHUNK 128
#define NCHUNK (IN_F / KCHUNK)   // 16 (even)

typedef _Float16 h16;
typedef __attribute__((ext_vector_type(16))) _Float16 v16h;
typedef __attribute__((ext_vector_type(8)))  _Float16 v8h;
typedef __attribute__((ext_vector_type(4)))  _Float16 v4h;
typedef __attribute__((ext_vector_type(8)))  float    v8f;

union V16H { v16h v; v8h h[2]; };

// 16B global -> LDS async copy (per-lane), tracked by ASYNCcnt.
__device__ __forceinline__ void async_cp16(void* lds_ptr, const void* gaddr) {
    uint32_t lds_off = (uint32_t)(uintptr_t)lds_ptr;   // low 32 bits of generic = LDS offset
    asm volatile("global_load_async_to_lds_b128 %0, %1, off"
                 :: "v"(lds_off), "v"(gaddr) : "memory");
}
__device__ __forceinline__ void wait_async0() {
    asm volatile("s_wait_asynccnt 0x0" ::: "memory");
}

// ---------------- conversion / setup kernels ----------------

__global__ void k_zero_counts(int* gcount) {
    if (threadIdx.x < HEADS) gcount[threadIdx.x] = 0;
}

// vectorized f32 -> f16 (4 elements per thread)
__global__ void k_cvt4(const float4* __restrict__ x, h16* __restrict__ xh, int n4) {
    int i = blockIdx.x * blockDim.x + threadIdx.x;
    if (i < n4) {
        float4 v = x[i];
        v4h o = { (h16)v.x, (h16)v.y, (h16)v.z, (h16)v.w };
        *(v4h*)(xh + (size_t)i * 4) = o;
    }
}

// weight (H, IN, HD) f32  ->  wh (H, HD, IN) f16  (B columns become K-contiguous)
__global__ void k_cvt_w(const float* __restrict__ w, h16* __restrict__ wh, int n) {
    int i = blockIdx.x * blockDim.x + threadIdx.x;
    if (i >= n) return;
    int h  = i / (IN_F * HD);
    int r  = i % (IN_F * HD);
    int in = r / HD;
    int d  = r % HD;
    wh[(size_t)h * HD * IN_F + (size_t)d * IN_F + in] = (h16)w[i];
}

// normalize centroids: one block per head, 256 threads (= HIDDEN)
__global__ void k_cent(const float* __restrict__ c, float* __restrict__ cn) {
    __shared__ float red[HIDDEN];
    int h = blockIdx.x, tid = threadIdx.x;
    float v = c[h * HIDDEN + tid];
    red[tid] = v * v;
    __syncthreads();
    for (int s = HIDDEN / 2; s > 0; s >>= 1) {
        if (tid < s) red[tid] += red[tid + s];
        __syncthreads();
    }
    float n = fmaxf(sqrtf(red[0]), 1e-12f);
    cn[h * HIDDEN + tid] = v / n;
}

// ---------------- routing: z GEMM (WMMA f16) + softmax + top-4 + list build ----------------

__global__ __launch_bounds__(256)
void k_route(const h16* __restrict__ xh, const h16* __restrict__ rh,
             const float* __restrict__ cn, const float* __restrict__ temp,
             int* __restrict__ gcount, int* __restrict__ glist,
             float* __restrict__ pscale)
{
    __shared__ h16   As[2][16][136];         // double-buffered A chunk: 16 tokens x 128 K
    __shared__ float zb[16][HIDDEN + 8];     // z tile
    __shared__ float lg[16][HEADS];          // logits

    const int t0   = blockIdx.x * 16;
    const int tid  = threadIdx.x;
    const int wv   = tid >> 5;               // wave 0..7 -> cols 32*wv .. 32*wv+31
    const int lane = tid & 31;
    const int lN   = lane & 15;
    const int hi   = lane >> 4;
    const int sr   = tid >> 4;               // staging row 0..15
    const int sc   = (tid & 15) * 8;         // staging 8-half chunk within 128

    v8f acc0 = {}, acc1 = {};
    const h16* bp0  = rh + (size_t)(wv * 32 + lN) * IN_F;        // column j = wv*32+lN
    const h16* bp1  = rh + (size_t)(wv * 32 + 16 + lN) * IN_F;   // column j+16
    const h16* arow = xh + (size_t)(t0 + sr) * IN_F + sc;

    auto stage = [&](int buf, int k0) {
        async_cp16(&As[buf][sr][sc], arow + k0);
    };
    auto compute = [&](int buf, int k0) {
#pragma unroll
        for (int ks = 0; ks < 4; ++ks) {
            const int kl = ks * 32;
            V16H a, b0, b1;
            a.h[0] = *(const v8h*)&As[buf][lN][kl + 16 * hi];      // K = 16*hi .. +7
            a.h[1] = *(const v8h*)&As[buf][lN][kl + 16 * hi + 8];  // K = 16*hi+8 .. +15
            const h16* p0 = bp0 + k0 + kl + 16 * hi;               // K = 16*hi .. +15
            const h16* p1 = bp1 + k0 + kl + 16 * hi;
            b0.h[0] = *(const v8h*)(p0);  b0.h[1] = *(const v8h*)(p0 + 8);
            b1.h[0] = *(const v8h*)(p1);  b1.h[1] = *(const v8h*)(p1 + 8);
            acc0 = __builtin_amdgcn_wmma_f32_16x16x32_f16(false, a.v, false, b0.v,
                                                          (short)0, acc0, false, false);
            acc1 = __builtin_amdgcn_wmma_f32_16x16x32_f16(false, a.v, false, b1.v,
                                                          (short)0, acc1, false, false);
        }
    };

    stage(0, 0);                                   // prologue: chunk 0 -> buf0
    wait_async0();
    __syncthreads();

    for (int c = 0; c < NCHUNK; c += 2) {          // 2x unrolled: fixed buffer roles
        stage(1, (c + 1) * KCHUNK);                // c+1 <= 15 always valid
        compute(0, c * KCHUNK);
        wait_async0();
        __syncthreads();
        if (c + 2 < NCHUNK) stage(0, (c + 2) * KCHUNK);
        compute(1, (c + 1) * KCHUNK);
        if (c + 2 < NCHUNK) wait_async0();
        __syncthreads();
    }

    // spill z tile to LDS:  C layout: VGPR r -> M = r + 8*hi, N = lN
#pragma unroll
    for (int r = 0; r < 8; ++r) {
        zb[r + 8 * hi][wv * 32 + lN]      = acc0[r];
        zb[r + 8 * hi][wv * 32 + 16 + lN] = acc1[r];
    }
    __syncthreads();

    // logits: thread = (token, head)
    {
        const int token = tid >> 4;
        const int hh    = tid & 15;
        float ss = 0.f, dot = 0.f;
        for (int j = 0; j < HIDDEN; ++j) {
            float zv = zb[token][j];
            ss  += zv * zv;
            dot += zv * cn[hh * HIDDEN + j];
        }
        float nrm = fmaxf(sqrtf(ss), 1e-12f);
        lg[token][hh] = (dot / nrm) * expf(temp[0]);
    }
    __syncthreads();

    // softmax + top-4 + renorm + list append : one thread per token
    if (tid < 16) {
        float pr[HEADS];
        float mx = -1e30f;
        for (int h = 0; h < HEADS; ++h) mx = fmaxf(mx, lg[tid][h]);
        float s = 0.f;
        for (int h = 0; h < HEADS; ++h) { pr[h] = expf(lg[tid][h] - mx); s += pr[h]; }

        int   ti[TOPK];
        float tp[TOPK];
        bool  used[HEADS] = {};
        for (int k = 0; k < TOPK; ++k) {
            int best = 0; float bv = -1e30f;
            for (int h = 0; h < HEADS; ++h)
                if (!used[h] && pr[h] > bv) { bv = pr[h]; best = h; }
            used[best] = true; ti[k] = best; tp[k] = bv;
        }
        float sum4  = (tp[0] + tp[1] + tp[2] + tp[3]) / s;
        float denom = sum4 + 1e-6f;
        int t = t0 + tid;
        for (int k = 0; k < TOPK; ++k) {
            float p = (tp[k] / s) / denom;
            pscale[t * TOPK + k] = p;
            int pos = atomicAdd(&gcount[ti[k]], 1);
            glist[ti[k] * NTOK + pos] = (t << 2) | k;
        }
    }
}

// ---------------- grouped head GEMM: only selected (token, head) pairs ----------------
// Block tile: 32 gathered rows x 128 cols. 8 waves = 2 (M) x 4 (N), each wave 16x32
// output (two accumulators, one shared A fragment -> 2 WMMAs per A-fragment load).

__global__ __launch_bounds__(256)
void k_head_gemm(const h16* __restrict__ xh, const h16* __restrict__ wh,
                 const float* __restrict__ bias, const int* __restrict__ gcount,
                 const int* __restrict__ glist, const float* __restrict__ pscale,
                 float* __restrict__ out)
{
    const int h    = blockIdx.y;
    const int tile = blockIdx.x;
    const int cnt  = gcount[h];
    if (tile * 32 >= cnt) return;

    __shared__ h16   As[2][32][136];         // double-buffered: 32 rows x 128 K
    __shared__ int   s_enc[32];
    __shared__ int   s_tok[32];
    __shared__ float s_p[32];

    const int tid  = threadIdx.x;
    const int wv   = tid >> 5;
    const int my   = wv >> 2;                // 0..1 : row half (16 rows)
    const int nx   = wv & 3;                 // 0..3 : col group (32 cols)
    const int lane = tid & 31;
    const int lN   = lane & 15;
    const int hi   = lane >> 4;
    const int sr   = tid >> 3;               // staging row 0..31
    const int sc   = (tid & 7) * 16;         // staging 16-half (32B) chunk within 128

    if (tid < 32) {
        int r   = tile * 32 + tid;
        int enc = (r < cnt) ? glist[h * NTOK + r] : -1;
        s_enc[tid] = enc;
        s_tok[tid] = (enc >= 0) ? (enc >> 2) : 0;
        s_p[tid]   = (enc >= 0) ? pscale[enc] : 0.f;
    }
    __syncthreads();

    const h16* wp0  = wh + (size_t)h * HD * IN_F + (size_t)(nx * 32 + lN) * IN_F;
    const h16* wp1  = wp0 + (size_t)16 * IN_F;
    const h16* arow = xh + (size_t)s_tok[sr] * IN_F + sc;

    v8f acc0 = {}, acc1 = {};

    auto stage = [&](int buf, int k0) {
        async_cp16(&As[buf][sr][sc],     arow + k0);
        async_cp16(&As[buf][sr][sc + 8], arow + k0 + 8);
    };
    auto compute = [&](int buf, int k0) {
#pragma unroll
        for (int ks = 0; ks < 4; ++ks) {
            const int kl = ks * 32;
            V16H a, b0, b1;
            a.h[0] = *(const v8h*)&As[buf][16 * my + lN][kl + 16 * hi];
            a.h[1] = *(const v8h*)&As[buf][16 * my + lN][kl + 16 * hi + 8];
            const h16* p0 = wp0 + k0 + kl + 16 * hi;
            const h16* p1 = wp1 + k0 + kl + 16 * hi;
            b0.h[0] = *(const v8h*)(p0);  b0.h[1] = *(const v8h*)(p0 + 8);
            b1.h[0] = *(const v8h*)(p1);  b1.h[1] = *(const v8h*)(p1 + 8);
            acc0 = __builtin_amdgcn_wmma_f32_16x16x32_f16(false, a.v, false, b0.v,
                                                          (short)0, acc0, false, false);
            acc1 = __builtin_amdgcn_wmma_f32_16x16x32_f16(false, a.v, false, b1.v,
                                                          (short)0, acc1, false, false);
        }
    };

    stage(0, 0);                                   // prologue: chunk 0 -> buf0
    wait_async0();
    __syncthreads();

    for (int c = 0; c < NCHUNK; c += 2) {          // 2x unrolled: fixed buffer roles
        stage(1, (c + 1) * KCHUNK);
        compute(0, c * KCHUNK);
        wait_async0();
        __syncthreads();
        if (c + 2 < NCHUNK) stage(0, (c + 2) * KCHUNK);
        compute(1, (c + 1) * KCHUNK);
        if (c + 2 < NCHUNK) wait_async0();
        __syncthreads();
    }

    const float bn0 = bias[h * HD + nx * 32 + lN];
    const float bn1 = bias[h * HD + nx * 32 + 16 + lN];
#pragma unroll
    for (int r = 0; r < 8; ++r) {
        const int M   = 16 * my + r + 8 * hi;      // row within 32-row tile
        const int enc = s_enc[M];
        if (enc >= 0) {
            int t = enc >> 2, k = enc & 3;
            float* op = out + (size_t)t * (TOPK * HD) + k * HD + nx * 32;
            op[lN]      = (acc0[r] + bn0) * s_p[M];
            op[16 + lN] = (acc1[r] + bn1) * s_p[M];
        }
    }
}

// ---------------- launcher ----------------

extern "C" void kernel_launch(void* const* d_in, const int* in_sizes, int n_in,
                              void* d_out, int out_size, void* d_ws, size_t ws_size,
                              hipStream_t stream) {
    const float* x        = (const float*)d_in[0];   // (4,2048,2048)
    const float* router_w = (const float*)d_in[1];   // (256,2048)
    const float* cent     = (const float*)d_in[2];   // (16,256)
    const float* temp     = (const float*)d_in[3];   // (1,)
    const float* weight   = (const float*)d_in[4];   // (16,2048,128)
    const float* bias     = (const float*)d_in[5];   // (16,128)
    float*       out      = (float*)d_out;           // (4,2048,512)

    char*  ws  = (char*)d_ws;
    size_t off = 0;
    auto alloc = [&](size_t bytes) -> void* {
        void* p = ws + off;
        off += (bytes + 255) & ~(size_t)255;
        return p;
    };

    h16*   xh     = (h16*)  alloc((size_t)NTOK * IN_F * sizeof(h16));       // 32 MB
    h16*   wh     = (h16*)  alloc((size_t)HEADS * HD * IN_F * sizeof(h16)); // 8 MB
    h16*   rh     = (h16*)  alloc((size_t)HIDDEN * IN_F * sizeof(h16));     // 1 MB
    float* cn     = (float*)alloc((size_t)HEADS * HIDDEN * sizeof(float));
    int*   gcount = (int*)  alloc((size_t)HEADS * sizeof(int));
    int*   glist  = (int*)  alloc((size_t)HEADS * NTOK * sizeof(int));      // 512 KB
    float* pscale = (float*)alloc((size_t)NTOK * TOPK * sizeof(float));     // 128 KB
    (void)ws_size; (void)in_sizes; (void)n_in; (void)out_size;

    k_zero_counts<<<1, 32, 0, stream>>>(gcount);

    {
        int n4 = NTOK * IN_F / 4;
        k_cvt4<<<(n4 + 255) / 256, 256, 0, stream>>>((const float4*)x, xh, n4);
    }
    {
        int n = HEADS * IN_F * HD;
        k_cvt_w<<<(n + 255) / 256, 256, 0, stream>>>(weight, wh, n);
    }
    {
        int n4 = HIDDEN * IN_F / 4;
        k_cvt4<<<(n4 + 255) / 256, 256, 0, stream>>>((const float4*)router_w, rh, n4);
    }
    k_cent<<<HEADS, HIDDEN, 0, stream>>>(cent, cn);

    k_route<<<NTOK / 16, 256, 0, stream>>>(xh, rh, cn, temp, gcount, glist, pscale);

    dim3 grid(NTOK / 32, HEADS, 1);
    k_head_gemm<<<grid, 256, 0, stream>>>(xh, wh, bias, gcount, glist, pscale, out);
}
